// GCNEncoder_3470333575319
// MI455X (gfx1250) — compile-verified, matrix operand order checked by reading, not run on hardware
//
#include <hip/hip_runtime.h>
#include <hip/hip_bf16.h>

#define N_NODES 50000
#define N_EDGES 800000
#define IN_CH   128
#define HID     128
#define OUT_CH  64

typedef __attribute__((ext_vector_type(2)))  float    v2f;
typedef __attribute__((ext_vector_type(8)))  float    v8f;
typedef __attribute__((ext_vector_type(16))) _Float16 v16h;

#if __has_builtin(__builtin_amdgcn_wmma_f32_16x16x4_f32)
#define HAVE_WMMA_F32 1
#else
#define HAVE_WMMA_F32 0
#endif

static __device__ __forceinline__ void atomic_add_f32(float* p, float v) {
    __hip_atomic_fetch_add(p, v, __ATOMIC_RELAXED, __HIP_MEMORY_SCOPE_AGENT);
}

// Async global->LDS 16-byte copy (gfx1250, ASYNCcnt-tracked).
// ldsoff = workgroup-relative LDS byte address, gsrc = 64-bit global address.
static __device__ __forceinline__ void async_copy_b128(unsigned ldsoff, const float* gsrc) {
    asm volatile("global_load_async_to_lds_b128 %0, %1, off"
                 :: "v"(ldsoff), "v"(gsrc)
                 : "memory");
}
static __device__ __forceinline__ void wait_asynccnt0() {
    asm volatile("s_wait_asynccnt 0x0" ::: "memory");
}

// ---------------- degree / normalization ----------------

__global__ void k_init_deg(float* __restrict__ deg) {
    int i = blockIdx.x * blockDim.x + threadIdx.x;
    if (i < N_NODES) deg[i] = 1.0f;  // self loop contributes 1
}

__global__ void k_edge_deg(const int* __restrict__ col, float* __restrict__ deg) {
    int e = blockIdx.x * blockDim.x + threadIdx.x;
    if (e < N_EDGES) atomic_add_f32(&deg[col[e]], 1.0f);
}

__global__ void k_dinv(float* __restrict__ deg) {
    int i = blockIdx.x * blockDim.x + threadIdx.x;
    if (i < N_NODES) deg[i] = rsqrtf(deg[i]);  // deg >= 1 always (self loops)
}

// ---------------- dense transform: Y[N, OUTC] = X[N, INC] @ W[INC, OUTC] ----------------
// Block handles 16 rows; OUTC/16 waves each own one 16-wide column tile.
// The 16xINC A tile is staged once into LDS with async b128 copies (the tile is
// one contiguous 16*INC*4-byte block), then every wave feeds WMMA from LDS.

template <int INC, int OUTC>
__global__ void k_gemm_wmma(const float* __restrict__ X,
                            const float* __restrict__ W,
                            float* __restrict__ Y) {
    __shared__ __align__(16) float tile[16 * INC];

    const int tid      = threadIdx.x;
    const int nthreads = (OUTC / 16) * 32;
    const int m0       = blockIdx.x * 16;

    // ---- stage A tile: contiguous copy of 16*INC floats via async DMA ----
    const int nchunks = (16 * INC) / 4;  // float4 chunks
    for (int ch = tid; ch < nchunks; ch += nthreads) {
        const float* gsrc = X + (size_t)m0 * INC + (size_t)ch * 4;
        unsigned ldsoff = (unsigned)(size_t)(&tile[ch * 4]);  // low 32 bits = LDS offset
        async_copy_b128(ldsoff, gsrc);
    }
    wait_asynccnt0();   // this wave's async copies done
    __syncthreads();    // whole block's copies visible

    const int lane = threadIdx.x & 31;
    const int wave = threadIdx.x >> 5;
    const int l    = lane & 15;   // row (A) / col (B,C)
    const int half = lane >> 4;   // lane group 0/1
    const int n0   = wave * 16;

    const float* __restrict__ arow = &tile[l * INC];

    v8f acc = {};

#if HAVE_WMMA_F32
    // V_WMMA_F32_16X16X4_F32. A (16x4 f32): lanes 0-15 hold K{0,1}, lanes 16-31 K{2,3};
    // B (4x16) mirrors with N across lanes.
    for (int k0 = 0; k0 < INC; k0 += 4) {
        const int ka = k0 + half * 2;
        v2f a = *(const v2f*)(arow + ka);          // ds_load_b64
        v2f b;
        b.x = W[(size_t)ka * OUTC + n0 + l];
        b.y = W[(size_t)(ka + 1) * OUTC + n0 + l];
        acc = __builtin_amdgcn_wmma_f32_16x16x4_f32(
            /*neg_a=*/false, a, /*neg_b=*/false, b,
            /*c_mod=*/(short)0, acc, /*reuse_a=*/false, /*reuse_b=*/false);
    }
#else
    // Fallback: V_WMMA_F32_16X16X32_F16, converting on load.
    for (int k0 = 0; k0 < INC; k0 += 32) {
        v16h a, b;
#pragma unroll
        for (int v = 0; v < 8; ++v) {
            const int kb = k0 + ((v < 4) ? 2 * v : 16 + 2 * (v - 4)) + half * 8;
            a[2 * v]     = (_Float16)arow[kb];
            a[2 * v + 1] = (_Float16)arow[kb + 1];
            b[2 * v]     = (_Float16)W[(size_t)kb * OUTC + n0 + l];
            b[2 * v + 1] = (_Float16)W[(size_t)(kb + 1) * OUTC + n0 + l];
        }
        acc = __builtin_amdgcn_wmma_f32_16x16x32_f16(
            false, a, false, b, (short)0, acc, false, false);
    }
#endif

    // C/D layout: VGPR r -> M = r + 8*half, N = l.
    float* __restrict__ ycol = Y + n0 + l;
#pragma unroll
    for (int r = 0; r < 8; ++r)
        ycol[(size_t)(m0 + r + half * 8) * OUTC] = acc[r];
}

// ---------------- self-loop message + bias init ----------------
// agg[i,c] = dinv[i]^2 * h[i,c] + bias[c]

template <int C>
__global__ void k_self_bias(const float* __restrict__ h,
                            const float* __restrict__ dinv,
                            const float* __restrict__ bias,
                            float* __restrict__ out) {
    int idx = blockIdx.x * blockDim.x + threadIdx.x;
    if (idx >= N_NODES * C) return;
    int i = idx / C;
    int c = idx & (C - 1);
    float dv = dinv[i];
    out[idx] = dv * dv * h[idx] + bias[c];
}

// ---------------- edge scatter: agg[col] += dinv[row]*dinv[col] * h[row] ----------------
// C/4 threads per edge, each does a float4 gather + 4 float atomics.

template <int C>
__global__ void k_scatter(const int* __restrict__ row,
                          const int* __restrict__ col,
                          const float* __restrict__ dinv,
                          const float* __restrict__ h,
                          float* __restrict__ agg) {
    const int TPE = C / 4;  // threads per edge (32 or 16)
    int gid = blockIdx.x * blockDim.x + threadIdx.x;
    int e = gid / TPE;
    if (e >= N_EDGES) return;
    int c = (gid & (TPE - 1)) * 4;

    int r = row[e];
    int t = col[e];
    float nrm = dinv[r] * dinv[t];

    const float4 v = *(const float4*)(h + (size_t)r * C + c);
    float* dst = agg + (size_t)t * C + c;
    atomic_add_f32(dst + 0, nrm * v.x);
    atomic_add_f32(dst + 1, nrm * v.y);
    atomic_add_f32(dst + 2, nrm * v.z);
    atomic_add_f32(dst + 3, nrm * v.w);
}

__global__ void k_relu(float* __restrict__ a, int n) {
    int idx = blockIdx.x * blockDim.x + threadIdx.x;
    if (idx < n) a[idx] = fmaxf(a[idx], 0.0f);
}

// ---------------- launch ----------------

extern "C" void kernel_launch(void* const* d_in, const int* in_sizes, int n_in,
                              void* d_out, int out_size, void* d_ws, size_t ws_size,
                              hipStream_t stream) {
    const float* x   = (const float*)d_in[0];
    const int*   ei  = (const int*)d_in[1];
    const int*   row = ei;             // source
    const int*   col = ei + N_EDGES;   // target
    const float* W1  = (const float*)d_in[2];
    const float* b1  = (const float*)d_in[3];
    const float* W2  = (const float*)d_in[4];
    const float* b2  = (const float*)d_in[5];
    float*       out = (float*)d_out;

    float* dinv = (float*)d_ws;                       // N
    float* h1   = dinv + N_NODES;                     // N*128
    float* agg1 = h1 + (size_t)N_NODES * HID;         // N*128
    float* h2   = agg1 + (size_t)N_NODES * HID;       // N*64

    const int TB = 256;

    // normalization
    k_init_deg<<<(N_NODES + TB - 1) / TB, TB, 0, stream>>>(dinv);
    k_edge_deg<<<(N_EDGES + TB - 1) / TB, TB, 0, stream>>>(col, dinv);
    k_dinv<<<(N_NODES + TB - 1) / TB, TB, 0, stream>>>(dinv);

    // layer 1: transform, self+bias, scatter, relu
    k_gemm_wmma<IN_CH, HID><<<N_NODES / 16, (HID / 16) * 32, 0, stream>>>(x, W1, h1);
    k_self_bias<HID><<<(N_NODES * HID + TB - 1) / TB, TB, 0, stream>>>(h1, dinv, b1, agg1);
    k_scatter<HID><<<(N_EDGES * (HID / 4) + TB - 1) / TB, TB, 0, stream>>>(row, col, dinv, h1, agg1);
    k_relu<<<(N_NODES * HID + TB - 1) / TB, TB, 0, stream>>>(agg1, N_NODES * HID);

    // layer 2: transform, self+bias (into d_out), scatter (into d_out)
    k_gemm_wmma<HID, OUT_CH><<<N_NODES / 16, (OUT_CH / 16) * 32, 0, stream>>>(agg1, W2, h2);
    k_self_bias<OUT_CH><<<(N_NODES * OUT_CH + TB - 1) / TB, TB, 0, stream>>>(h2, dinv, b2, out);
    k_scatter<OUT_CH><<<(N_EDGES * (OUT_CH / 4) + TB - 1) / TB, TB, 0, stream>>>(row, col, dinv, h2, out);
}
